// Qwen3AttentionFused_34557306863645
// MI455X (gfx1250) — matrix operand order repaired and checked by reading, not compiled, not run on hardware
//
#include <hip/hip_runtime.h>
#include <hip/hip_bf16.h>

typedef __bf16 bf16;
typedef bf16  v8bf  __attribute__((ext_vector_type(8)));
typedef bf16  v16bf __attribute__((ext_vector_type(16)));
typedef float v8f   __attribute__((ext_vector_type(8)));
typedef unsigned int u32x4 __attribute__((ext_vector_type(4)));
typedef int i32x8 __attribute__((ext_vector_type(8)));
typedef int i32x4 __attribute__((ext_vector_type(4)));

#define HIDc 2048
#define NHc  16
#define NKVc 8
#define HDc  128
#define NQKVc 4096   // (NH + 2*NKV) * HD
#define SCALEc 0.08838834764831845f  // 128^-0.5

static __device__ __forceinline__ v16bf cat8(v8bf lo, v8bf hi) {
    return __builtin_shufflevector(lo, hi, 0,1,2,3,4,5,6,7,8,9,10,11,12,13,14,15);
}

// ---------------------------------------------------------------------------
// TDM issue: DMA a rows x cols bf16 tile (row stride = strideElems) from
// global memory into LDS at lds_off. 2D tensor descriptor, data_size=2B.
// One issue per wave; tracked with TENSORcnt.
// ---------------------------------------------------------------------------
static __device__ __forceinline__ void tdm_load_tile_bf16(const bf16* gsrc,
                                                          unsigned lds_off,
                                                          int rows, int cols,
                                                          int strideElems,
                                                          int tensorRows) {
    unsigned long long ga = (unsigned long long)(uintptr_t)gsrc;
    unsigned td0 = (unsigned)strideElems;     // tensor dim0 (elements)
    unsigned td1 = (unsigned)tensorRows;      // tensor dim1 (rows)

    u32x4 g0;
    g0[0] = 1u;                                           // count=1, user mode
    g0[1] = lds_off;                                      // lds_addr (bytes)
    g0[2] = (unsigned)(ga & 0xffffffffu);                 // global_addr[31:0]
    g0[3] = (unsigned)((ga >> 32) & 0x01ffffffu)          // global_addr[56:32]
            | (2u << 30);                                 // type = 2 ("image")

    i32x8 g1;
    g1[0] = (int)(1u << 16);                              // data_size = 1 -> 2 bytes
    g1[1] = (int)((td0 & 0xffffu) << 16);                 // tensor_dim0[15:0]
    g1[2] = (int)((td0 >> 16) | ((td1 & 0xffffu) << 16)); // dim0[31:16] | dim1[15:0]
    g1[3] = (int)((td1 >> 16) | ((unsigned)cols << 16));  // dim1[31:16] | tile_dim0
    g1[4] = rows;                                         // tile_dim1 (tile_dim2=0)
    g1[5] = (int)td0;                                     // tensor_dim0_stride[31:0]
    g1[6] = 0;                                            // stride0[47:32], stride1 lo
    g1[7] = 0;

    i32x4 gz4 = {0, 0, 0, 0};
    i32x8 gz8 = {0, 0, 0, 0, 0, 0, 0, 0};
    __builtin_amdgcn_tensor_load_to_lds(g0, g1, gz4, gz4, gz8, 0);
}

// ---------------------------------------------------------------------------
// fp32 -> bf16 elementwise conversion
// ---------------------------------------------------------------------------
__global__ void cvt_f32_bf16_kernel(const float* __restrict__ in, bf16* __restrict__ out, int n) {
    int i = blockIdx.x * blockDim.x + threadIdx.x;
    if (i < n) out[i] = (bf16)in[i];
}

// ---------------------------------------------------------------------------
// Row RMSNorm of hidden states (S x 2048 fp32) -> bf16
// ---------------------------------------------------------------------------
__global__ __launch_bounds__(256) void rmsnorm_kernel(const float* __restrict__ X,
                                                      const float* __restrict__ w,
                                                      bf16* __restrict__ out) {
    int s = blockIdx.x, tid = threadIdx.x;
    const float* row = X + (size_t)s * HIDc;
    float ss = 0.f;
    for (int i = tid; i < HIDc; i += 256) { float v = row[i]; ss += v * v; }
    for (int off = 16; off; off >>= 1) ss += __shfl_xor(ss, off);
    __shared__ float red[8];
    if ((tid & 31) == 0) red[tid >> 5] = ss;
    __syncthreads();
    float tot = 0.f;
    for (int i = 0; i < 8; i++) tot += red[i];
    float sc = rsqrtf(tot * (1.f / HIDc) + 1e-6f);
    for (int i = tid; i < HIDc; i += 256)
        out[(size_t)s * HIDc + i] = (bf16)(row[i] * sc * w[i]);
}

// ---------------------------------------------------------------------------
// bf16 GEMM with TDM double buffering: C[M,N] = A[M,K] * W[N,K]^T, fp32 out.
// BM=BN=128, BK=32. Wave 0 issues tensor_load_to_lds for the next K-slab
// while all 8 waves run WMMA on the current slab.
// ---------------------------------------------------------------------------
__global__ __launch_bounds__(256) void gemm_bf16_nt(const bf16* __restrict__ A,
                                                    const bf16* __restrict__ W,
                                                    float* __restrict__ C,
                                                    int M, int N, int K) {
    __shared__ bf16 As[2][128][32];
    __shared__ bf16 Ws[2][128][32];
    int tid = threadIdx.x, wave = tid >> 5, lane = tid & 31;
    int bm = blockIdx.y * 128, bn = blockIdx.x * 128;
    int wm = (wave & 3) * 32, wn = (wave >> 2) * 64;
    int col = lane & 15, half = lane >> 4;
    int koff = half * 8, kb = half * 8;

    unsigned asOff[2], wsOff[2];
    for (int b = 0; b < 2; b++) {
        asOff[b] = (unsigned)(uintptr_t)(void*)&As[b][0][0];
        wsOff[b] = (unsigned)(uintptr_t)(void*)&Ws[b][0][0];
    }

    v8f acc[2][4] = {};

    int nk = K >> 5;
    if (wave == 0) {
        tdm_load_tile_bf16(A + (size_t)bm * K, asOff[0], 128, 32, K, M);
        tdm_load_tile_bf16(W + (size_t)bn * K, wsOff[0], 128, 32, K, N);
    }

    for (int ki = 0; ki < nk; ki++) {
        int cur = ki & 1;
        if (wave == 0) __builtin_amdgcn_s_wait_tensorcnt(0);
        __syncthreads();                   // slab `cur` ready for everyone
        if (wave == 0 && ki + 1 < nk) {    // prefetch next slab into other buffer
            int k1 = (ki + 1) << 5;
            tdm_load_tile_bf16(A + (size_t)bm * K + k1, asOff[cur ^ 1], 128, 32, K, M);
            tdm_load_tile_bf16(W + (size_t)bn * K + k1, wsOff[cur ^ 1], 128, 32, K, N);
        }

        v16bf af[2], bfrag[4];
        for (int i = 0; i < 2; i++) {
            int m = wm + i * 16 + col;
            v8bf lo = *(const v8bf*)&As[cur][m][koff];
            v8bf hi = *(const v8bf*)&As[cur][m][16 + koff];
            af[i] = cat8(lo, hi);
        }
        for (int j = 0; j < 4; j++) {
            int n = wn + j * 16 + col;
            v8bf lo = *(const v8bf*)&Ws[cur][n][kb];
            v8bf hi = *(const v8bf*)&Ws[cur][n][kb + 16];
            bfrag[j] = cat8(lo, hi);
        }
        for (int i = 0; i < 2; i++)
            for (int j = 0; j < 4; j++)
                acc[i][j] = __builtin_amdgcn_wmma_f32_16x16x32_bf16(
                    false, af[i], false, bfrag[j], (short)0, acc[i][j], false, false);
    }

    for (int i = 0; i < 2; i++)
        for (int j = 0; j < 4; j++)
            for (int r = 0; r < 8; r++) {
                int m = bm + wm + i * 16 + half * 8 + r;
                int n = bn + wn + j * 16 + col;
                C[(size_t)m * N + n] = acc[i][j][r];
            }
}

// ---------------------------------------------------------------------------
// q/k head RMSNorm + RoPE; writes Qb[NH][S][HD], Kb[NKV][S][HD] bf16,
// and V transposed Vt[NKV][HD][S] bf16.
// ---------------------------------------------------------------------------
__global__ __launch_bounds__(256) void ropenorm_kernel(const float* __restrict__ qkv,
                                                       const int* __restrict__ pos,
                                                       const float* __restrict__ qw,
                                                       const float* __restrict__ kw,
                                                       bf16* __restrict__ Qb,
                                                       bf16* __restrict__ Kb,
                                                       bf16* __restrict__ Vt, int S) {
    int s = blockIdx.x;
    int tid = threadIdx.x, wave = tid >> 5, lane = tid & 31;
    float p = (float)pos[s];
    __shared__ float buf[8][128];
    const float* base = qkv + (size_t)s * NQKVc;

    for (int hh = wave; hh < NHc + NKVc; hh += 8) {
        bool isQ = hh < NHc;
        const float* hp = base + (isQ ? hh * HDc : NHc * HDc + (hh - NHc) * HDc);
        const float* nw = isQ ? qw : kw;
        float x[4]; float ss = 0.f;
        for (int i = 0; i < 4; i++) { x[i] = hp[lane * 4 + i]; ss += x[i] * x[i]; }
        for (int off = 16; off; off >>= 1) ss += __shfl_xor(ss, off);
        float sc = rsqrtf(ss * (1.f / HDc) + 1e-6f);
        for (int i = 0; i < 4; i++)
            buf[wave][lane * 4 + i] = x[i] * sc * nw[lane * 4 + i];
        asm volatile("s_wait_dscnt 0" ::: "memory");
        bf16* dst = isQ ? (Qb + ((size_t)hh * S + s) * HDc)
                        : (Kb + ((size_t)(hh - NHc) * S + s) * HDc);
        for (int i = 0; i < 4; i++) {
            int dd = lane * 4 + i;
            int j = (dd < 64) ? dd : dd - 64;
            float invf = __powf(10000.f, -(float)(2 * j) * (1.f / HDc));
            float fr = p * invf;
            float sn, cs; __sincosf(fr, &sn, &cs);
            float x1 = buf[wave][j], x2 = buf[wave][j + 64];
            float o = (dd < 64) ? (x1 * cs - x2 * sn) : (x2 * cs + x1 * sn);
            dst[dd] = (bf16)o;
        }
        asm volatile("s_wait_dscnt 0" ::: "memory");
    }

    // V: copy transposed
    for (int i = 0; i < 4; i++) {
        int idx = tid * 4 + i;                // 0..1023
        int vh = idx >> 7, d = idx & 127;
        Vt[((size_t)(vh * HDc + d)) * S + s] = (bf16)base[(NHc + NKVc) * HDc + idx];
    }
}

// ---------------------------------------------------------------------------
// Flash-style causal attention. Grid: (S/128, NH). One wave = 16 queries.
// Qb[NH][S][HD], Kb[NKV][S][HD], Vt[NKV][HD][S]; Ob[S][NH*HD] bf16.
// ---------------------------------------------------------------------------
__global__ __launch_bounds__(256) void attn_kernel(const bf16* __restrict__ Qb,
                                                   const bf16* __restrict__ Kb,
                                                   const bf16* __restrict__ Vt,
                                                   bf16* __restrict__ Ob, int S) {
    int h = blockIdx.y;
    int kv = h >> 1;                           // GQA rep = NH/NKV = 2
    int tid = threadIdx.x, wave = tid >> 5, lane = tid & 31;
    int q0 = blockIdx.x * 128 + wave * 16;
    int col = lane & 15, half = lane >> 4;
    int koff = half * 8, kb16 = half * 8;

    __shared__ bf16 Ps[8][16][32];

    // Preload Q A-fragments (4 chunks of K=32 along HD=128)
    const bf16* qrow = Qb + ((size_t)h * S + q0 + col) * HDc;
    v16bf qa[4];
    for (int c = 0; c < 4; c++) {
        v8bf lo = *(const v8bf*)(qrow + c * 32 + koff);
        v8bf hi = *(const v8bf*)(qrow + c * 32 + 16 + koff);
        qa[c] = cat8(lo, hi);
    }

    float mrow[8], lrow[8];
    for (int r = 0; r < 8; r++) { mrow[r] = -3.0e38f; lrow[r] = 0.f; }
    v8f ov[8] = {};

    int jmax = (q0 + 15) >> 5;                  // causal bound on 32-key tiles
    for (int jt = 0; jt <= jmax; jt++) {
        int t0 = jt * 32;

        // scores: 16 x 32 in two 16x16 fragments
        v8f sf[2] = {};
        for (int nt = 0; nt < 2; nt++) {
            const bf16* krow = Kb + ((size_t)kv * S + t0 + nt * 16 + col) * HDc + kb16;
            for (int c = 0; c < 4; c++) {
                v8bf lo = *(const v8bf*)(krow + c * 32);
                v8bf hi = *(const v8bf*)(krow + c * 32 + 16);
                v16bf kf = cat8(lo, hi);
                sf[nt] = __builtin_amdgcn_wmma_f32_16x16x32_bf16(
                    false, qa[c], false, kf, (short)0, sf[nt], false, false);
            }
        }

        // scale + causal mask
        float sv0[8], sv1[8];
        for (int r = 0; r < 8; r++) {
            int mg = q0 + half * 8 + r;
            int tA = t0 + col, tB = t0 + 16 + col;
            sv0[r] = (tA > mg) ? -3.0e38f : sf[0][r] * SCALEc;
            sv1[r] = (tB > mg) ? -3.0e38f : sf[1][r] * SCALEc;
        }

        // online softmax: row max across 16-lane half
        float rmax[8];
        for (int r = 0; r < 8; r++) rmax[r] = fmaxf(sv0[r], sv1[r]);
        for (int off = 1; off < 16; off <<= 1)
            for (int r = 0; r < 8; r++)
                rmax[r] = fmaxf(rmax[r], __shfl_xor(rmax[r], off));

        float corr[8], p0[8], p1[8], rs[8];
        for (int r = 0; r < 8; r++) {
            float mnew = fmaxf(mrow[r], rmax[r]);
            corr[r] = __expf(mrow[r] - mnew);
            p0[r] = __expf(sv0[r] - mnew);
            p1[r] = __expf(sv1[r] - mnew);
            rs[r] = p0[r] + p1[r];
            mrow[r] = mnew;
        }
        for (int off = 1; off < 16; off <<= 1)
            for (int r = 0; r < 8; r++)
                rs[r] += __shfl_xor(rs[r], off);
        for (int r = 0; r < 8; r++) lrow[r] = lrow[r] * corr[r] + rs[r];
        for (int dt = 0; dt < 8; dt++)
            for (int r = 0; r < 8; r++) ov[dt][r] *= corr[r];

        // relayout P (C-layout) -> A-fragment via per-wave LDS
        for (int r = 0; r < 8; r++) {
            Ps[wave][half * 8 + r][col]      = (bf16)p0[r];
            Ps[wave][half * 8 + r][16 + col] = (bf16)p1[r];
        }
        asm volatile("s_wait_dscnt 0" ::: "memory");
        v8bf plo = *(const v8bf*)&Ps[wave][col][koff];
        v8bf phi = *(const v8bf*)&Ps[wave][col][16 + koff];
        v16bf pa = cat8(plo, phi);

        // P(16x32) x V(32x128): 8 output d-tiles
        for (int dt = 0; dt < 8; dt++) {
            const bf16* vrow = Vt + ((size_t)(kv * HDc + dt * 16 + col)) * S + t0 + kb16;
            v8bf lo = *(const v8bf*)vrow;
            v8bf hi = *(const v8bf*)(vrow + 16);
            v16bf vf = cat8(lo, hi);
            ov[dt] = __builtin_amdgcn_wmma_f32_16x16x32_bf16(
                false, pa, false, vf, (short)0, ov[dt], false, false);
        }
    }

    // normalize and store bf16 into Ob[S][NH*HD]
    for (int dt = 0; dt < 8; dt++)
        for (int r = 0; r < 8; r++) {
            int mg = q0 + half * 8 + r;
            float v = ov[dt][r] / lrow[r];
            Ob[(size_t)mg * (NHc * HDc) + h * HDc + dt * 16 + col] = (bf16)v;
        }
}

// ---------------------------------------------------------------------------
extern "C" void kernel_launch(void* const* d_in, const int* in_sizes, int n_in,
                              void* d_out, int out_size, void* d_ws, size_t ws_size,
                              hipStream_t stream) {
    const int*   positions = (const int*)d_in[0];
    const float* hidden    = (const float*)d_in[1];
    const float* lnw       = (const float*)d_in[2];
    const float* w_qkv     = (const float*)d_in[3];
    const float* w_o       = (const float*)d_in[4];
    const float* qnw       = (const float*)d_in[5];
    const float* knw       = (const float*)d_in[6];
    float*       out       = (float*)d_out;
    const int S = in_sizes[0];   // 2048

    char* p = (char*)d_ws;
    auto alloc = [&](size_t bytes) -> char* {
        char* r = p; p += (bytes + 255) & ~(size_t)255; return r;
    };
    bf16*  hnb   = (bf16*)alloc((size_t)S * HIDc * 2);
    bf16*  wqkvb = (bf16*)alloc((size_t)NQKVc * HIDc * 2);
    bf16*  wob   = (bf16*)alloc((size_t)HIDc * (NHc * HDc) * 2);
    float* qkv   = (float*)alloc((size_t)S * NQKVc * 4);
    bf16*  Qb    = (bf16*)alloc((size_t)NHc * S * HDc * 2);
    bf16*  Kb    = (bf16*)alloc((size_t)NKVc * S * HDc * 2);
    bf16*  Vt    = (bf16*)alloc((size_t)NKVc * HDc * S * 2);
    bf16*  Ob    = (bf16*)alloc((size_t)S * NHc * HDc * 2);
    (void)ws_size;

    int n1 = NQKVc * HIDc;
    cvt_f32_bf16_kernel<<<(n1 + 255) / 256, 256, 0, stream>>>(w_qkv, wqkvb, n1);
    int n2 = HIDc * NHc * HDc;
    cvt_f32_bf16_kernel<<<(n2 + 255) / 256, 256, 0, stream>>>(w_o, wob, n2);

    rmsnorm_kernel<<<S, 256, 0, stream>>>(hidden, lnw, hnb);

    dim3 g1(NQKVc / 128, S / 128);
    gemm_bf16_nt<<<g1, 256, 0, stream>>>(hnb, wqkvb, qkv, S, NQKVc, HIDc);

    ropenorm_kernel<<<S, 256, 0, stream>>>(qkv, positions, qnw, knw, Qb, Kb, Vt, S);

    dim3 ga(S / 128, NHc);
    attn_kernel<<<ga, 256, 0, stream>>>(Qb, Kb, Vt, Ob, S);

    dim3 g2(HIDc / 128, S / 128);
    gemm_bf16_nt<<<g2, 256, 0, stream>>>(Ob, wob, out, S, HIDc, HIDc);
}
